// WindowAttention_5574867550731
// MI455X (gfx1250) — compile-verified, matrix operand order checked by reading, not run on hardware
//
#include <hip/hip_runtime.h>
#include <hip/hip_bf16.h>

// ---------------------------------------------------------------------------
// Swin window attention, fused, gfx1250 (CDNA5) wave32 WMMA f16 path.
// B operands pre-swizzled into WMMA fragment order -> b128 loads everywhere.
// ---------------------------------------------------------------------------

typedef __attribute__((ext_vector_type(16))) _Float16 v16h;
typedef __attribute__((ext_vector_type(8)))  float    v8f;

#define NTOK 49      // tokens per window
#define NPAD 64      // padded tokens (4 x 16 WMMA tiles)
#define CH   192     // channels
#define DH   32      // head dim
#define NH   6       // heads
#define NWIN 64      // windows per image (mask groups)

// ---- WMMA fragment helpers (ISA 7.12.2 wave32 layouts) --------------------
// A (16x32 f16, MxK): lane L holds row (L&15); lanes>=16 start at K offset 8.
// Contiguous 8-half runs -> ds_load_b128 pairs.
__device__ __forceinline__ v16h frag_a(const _Float16* __restrict__ base,
                                       int ld, int lane) {
  const int row = lane & 15;
  const int kb  = (lane & 16) ? 8 : 0;
  v16h a;
#pragma unroll
  for (int j = 0; j < 8; ++j) {
    const int kj = kb + ((j < 4) ? (2 * j) : (16 + 2 * (j - 4)));
    a[2 * j]     = base[row * ld + kj];
    a[2 * j + 1] = base[row * ld + kj + 1];
  }
  return a;
}

// B fragment from pre-swizzled storage: lane L's 16 halfs are contiguous.
__device__ __forceinline__ v16h frag_bp(const _Float16* __restrict__ packed,
                                        int lane) {
  return *(const v16h*)(packed + lane * 16);
}

// Inverse B-layout mapping: element slot for K-index k (0..31), and the
// lane-group bit. lane = n + ((k>>3)&1)*16 ; elem = (k&7) | ((k>>4)<<3).
__device__ __forceinline__ int b_elem(int k) { return (k & 7) | ((k >> 4) << 3); }
__device__ __forceinline__ int b_lgrp(int k) { return (k >> 3) & 1; }

__device__ __forceinline__ v8f wmma_f16(v16h a, v16h b, v8f c) {
  return __builtin_amdgcn_wmma_f32_16x16x32_f16(
      false, a, false, b, (short)0, c, false, false);
}

__device__ __forceinline__ float wave_max(float v) {
#pragma unroll
  for (int m = 16; m >= 1; m >>= 1) v = fmaxf(v, __shfl_xor(v, m, 32));
  return v;
}
__device__ __forceinline__ float wave_sum(float v) {
#pragma unroll
  for (int m = 16; m >= 1; m >>= 1) v += __shfl_xor(v, m, 32);
  return v;
}

// ---- Prep: weights f32 -> f16 in B-fragment order; gather rel-pos bias ----
// Packed layout: P[((nt*6 + ks)*32 + lane)*16 + elem], nt = column tile.
__device__ __forceinline__ void pack_w(const float* __restrict__ W,
                                       _Float16* __restrict__ P, int N,
                                       int tid, int nthr) {
  for (int i = tid; i < CH * N; i += nthr) {
    const int kk = i / N, n = i % N;
    const int nt = n >> 4, nc = n & 15;
    const int ks = kk >> 5, kl = kk & 31;
    P[((nt * 6 + ks) * 32 + (nc + b_lgrp(kl) * 16)) * 16 + b_elem(kl)] =
        (_Float16)W[i];
  }
}

__global__ void swin_prep_kernel(const float* __restrict__ Wq,
                                 const float* __restrict__ Wkv,
                                 const float* __restrict__ Wproj,
                                 const float* __restrict__ bias_table,
                                 const int*   __restrict__ rel_index,
                                 _Float16* __restrict__ WqP,
                                 _Float16* __restrict__ WkvP,
                                 _Float16* __restrict__ WprojP,
                                 float* __restrict__ biasG) {
  const int tid  = blockIdx.x * blockDim.x + threadIdx.x;
  const int nthr = gridDim.x * blockDim.x;
  pack_w(Wq, WqP, CH, tid, nthr);
  pack_w(Wkv, WkvP, 2 * CH, tid, nthr);
  pack_w(Wproj, WprojP, CH, tid, nthr);
  // biasG[h][q][k] = bias_table[rel_index[q*49+k]][h]
  for (int i = tid; i < NH * NTOK * NTOK; i += nthr) {
    const int h  = i / (NTOK * NTOK);
    const int qk = i % (NTOK * NTOK);
    biasG[i] = bias_table[rel_index[qk] * NH + h];
  }
}

// ---- Fused window attention: one workgroup (8 wave32) per window ----------
__launch_bounds__(256, 2)
__global__ void swin_attn_kernel(const float* __restrict__ xq,
                                 const float* __restrict__ mask,
                                 const float* __restrict__ bq,
                                 const float* __restrict__ bkv,
                                 const float* __restrict__ bproj,
                                 const _Float16* __restrict__ WqP,
                                 const _Float16* __restrict__ WkvP,
                                 const _Float16* __restrict__ WprojP,
                                 const float* __restrict__ biasG,
                                 float* __restrict__ out,
                                 int nwin_total) {
  extern __shared__ char smem[];
  _Float16* Xh  = (_Float16*)smem;        // [64][192] X row-major; reused as O
  _Float16* Qh  = Xh  + NPAD * CH;        // [64][192] scaled Q row-major (A op)
  _Float16* KtP = Qh  + NPAD * CH;        // K in B-frag order: [h][nt(tok)][lane][16]
  _Float16* VP  = KtP + NH * 4 * 512;     // V in B-frag order: [h][nt(dh)][ks(tok)][lane][16]
  _Float16* Ph  = VP  + NH * 4 * 512;     // [64][64] softmax probs (A operand)
  float*    Sf  = (float*)(Ph + NPAD * NPAD);   // [64][64] logits
  _Float16* Oh  = Xh;                     // alias: X dead after QKV GEMM

  const int tid   = threadIdx.x;
  const int lane  = tid & 31;
  const int wid   = tid >> 5;
  const int win   = blockIdx.x;
  const int wmask = win & (NWIN - 1);
  const float scale = 0.17677669529663687f;  // 32^-0.5

  // ---- prefetch next window's activations (global_prefetch_b8 hint) ----
  if (win + 1 < nwin_total) {
    const char* nx = (const char*)(xq + (size_t)(win + 1) * NTOK * CH);
    __builtin_prefetch(nx + tid * 128, 0, 0);          // first 32 KB
    __builtin_prefetch(nx + 32768 + tid * 16, 0, 0);   // tail
  }

  // ---- stage X into LDS as f16 (vectorized), zero-pad rows 49..63 ----
  const float* xw = xq + (size_t)win * NTOK * CH;
  const float4* xw4 = (const float4*)xw;
  for (int i = tid; i < (NTOK * CH) / 4; i += 256) {
    const float4 t4 = xw4[i];
    _Float16* d = Xh + i * 4;
    d[0] = (_Float16)t4.x; d[1] = (_Float16)t4.y;
    d[2] = (_Float16)t4.z; d[3] = (_Float16)t4.w;
  }
  for (int i = NTOK * CH + tid; i < NPAD * CH; i += 256)
    Xh[i] = (_Float16)0.0f;
  __syncthreads();

  // ---- fused QKV projection: [64,192] x [192,576], K = 6 x 32 ----
  for (int t = wid; t < 144; t += 8) {
    int mt, nt, isQ;
    const _Float16* Wb;
    if (t < 48) { mt = t / 12; nt = t % 12; Wb = WqP;  isQ = 1; }
    else { const int t2 = t - 48; mt = t2 / 24; nt = t2 % 24; Wb = WkvP; isQ = 0; }
    v8f acc = {};
#pragma unroll
    for (int ks = 0; ks < 6; ++ks) {
      v16h a = frag_a(Xh + mt * 16 * CH + ks * 32, CH, lane);
      v16h b = frag_bp(Wb + (nt * 6 + ks) * 512, lane);
      acc = wmma_f16(a, b, acc);
    }
    const int col  = lane & 15;
    const int rowb = (lane & 16) ? 8 : 0;
    const int gcol = nt * 16 + col;
    if (isQ) {
      const float bb = bq[gcol];
#pragma unroll
      for (int i = 0; i < 8; ++i)
        Qh[(mt * 16 + rowb + i) * CH + gcol] = (_Float16)((acc[i] + bb) * scale);
    } else if (gcol < CH) {
      // K: scatter into B-frag order; Kdim = head-local d, N = token
      const int h  = gcol >> 5, dl = gcol & 31;
      const int lg = b_lgrp(dl) * 16, e = b_elem(dl);
      const float bb = bkv[gcol];
#pragma unroll
      for (int i = 0; i < 8; ++i) {
        const int tok = mt * 16 + rowb + i;
        KtP[((h * 4 + (tok >> 4)) * 32 + (tok & 15) + lg) * 16 + e] =
            (_Float16)(acc[i] + bb);
      }
    } else {
      // V: scatter into B-frag order; Kdim = token, N = head-local d
      const int c = gcol - CH;
      const int h = c >> 5, dl = c & 31;
      const int nt2 = dl >> 4, n = dl & 15;
      const float bb = bkv[gcol];
#pragma unroll
      for (int i = 0; i < 8; ++i) {
        const int tok = mt * 16 + rowb + i;
        const int ks2 = tok >> 5, kl = tok & 31;
        VP[(((h * 2 + nt2) * 2 + ks2) * 32 + n + b_lgrp(kl) * 16) * 16 + b_elem(kl)] =
            (_Float16)(acc[i] + bb);
      }
    }
  }
  __syncthreads();

  // ---- per-head attention ----
  for (int h = 0; h < NH; ++h) {
    // S = Q_h . K_h^T : 16 tiles, K = 32 (one WMMA each)
    for (int t = wid; t < 16; t += 8) {
      const int mt = t >> 2, nt = t & 3;
      v16h a = frag_a(Qh + mt * 16 * CH + h * DH, CH, lane);
      v16h b = frag_bp(KtP + (h * 4 + nt) * 512, lane);
      v8f acc = {};
      acc = wmma_f16(a, b, acc);
      const int col  = lane & 15;
      const int rowb = (lane & 16) ? 8 : 0;
#pragma unroll
      for (int i = 0; i < 8; ++i)
        Sf[(mt * 16 + rowb + i) * NPAD + nt * 16 + col] = acc[i];
    }
    __syncthreads();

    // softmax: 8 waves x lane-parallel k, wave32 shuffle reductions
    for (int r = wid; r < NPAD; r += 8) {
      if (r < NTOK) {
        const float* bp = biasG + ((h * NTOK) + r) * NTOK;
        const float* mp = mask + ((size_t)wmask * NTOK + r) * NTOK;
        const int k1 = lane + 32;
        const float v0 = (lane < NTOK) ? Sf[r * NPAD + lane] + bp[lane] + mp[lane]
                                       : -3.0e38f;
        const float v1 = (k1 < NTOK) ? Sf[r * NPAD + k1] + bp[k1] + mp[k1]
                                     : -3.0e38f;
        const float m  = wave_max(fmaxf(v0, v1));
        const float e0 = (lane < NTOK) ? __expf(v0 - m) : 0.0f;
        const float e1 = (k1 < NTOK) ? __expf(v1 - m) : 0.0f;
        const float inv = 1.0f / wave_sum(e0 + e1);
        Ph[r * NPAD + lane] = (_Float16)(e0 * inv);   // k>=49 -> exact 0
        Ph[r * NPAD + k1]   = (_Float16)(e1 * inv);
      } else {
        Ph[r * NPAD + lane]      = (_Float16)0.0f;
        Ph[r * NPAD + lane + 32] = (_Float16)0.0f;
      }
    }
    __syncthreads();

    // O_h = P . V_h : 8 tiles (4x2), K = 64 = 2 x 32
    for (int t = wid; t < 8; t += 8) {
      const int mt = t >> 1, nt = t & 1;
      v8f acc = {};
#pragma unroll
      for (int ks = 0; ks < 2; ++ks) {
        v16h a = frag_a(Ph + mt * 16 * NPAD + ks * 32, NPAD, lane);
        v16h b = frag_bp(VP + ((h * 2 + nt) * 2 + ks) * 512, lane);
        acc = wmma_f16(a, b, acc);
      }
      const int col  = lane & 15;
      const int rowb = (lane & 16) ? 8 : 0;
#pragma unroll
      for (int i = 0; i < 8; ++i)
        Oh[(mt * 16 + rowb + i) * CH + h * DH + nt * 16 + col] = (_Float16)acc[i];
    }
    __syncthreads();
  }

  // ---- output projection: [64,192] x [192,192], rows<49 to HBM (f32) ----
  float* ow = out + (size_t)win * NTOK * CH;
  for (int t = wid; t < 48; t += 8) {
    const int mt = t / 12, nt = t % 12;
    v8f acc = {};
#pragma unroll
    for (int ks = 0; ks < 6; ++ks) {
      v16h a = frag_a(Oh + mt * 16 * CH + ks * 32, CH, lane);
      v16h b = frag_bp(WprojP + (nt * 6 + ks) * 512, lane);
      acc = wmma_f16(a, b, acc);
    }
    const int col  = lane & 15;
    const int rowb = (lane & 16) ? 8 : 0;
    const float bb = bproj[nt * 16 + col];
#pragma unroll
    for (int i = 0; i < 8; ++i) {
      const int r = mt * 16 + rowb + i;
      if (r < NTOK) ow[r * CH + nt * 16 + col] = acc[i] + bb;
    }
  }
}

extern "C" void kernel_launch(void* const* d_in, const int* in_sizes, int n_in,
                              void* d_out, int out_size, void* d_ws, size_t ws_size,
                              hipStream_t stream) {
  const float* xq         = (const float*)d_in[0];
  const float* mask       = (const float*)d_in[1];
  const float* Wq         = (const float*)d_in[2];
  const float* bq         = (const float*)d_in[3];
  const float* Wkv        = (const float*)d_in[4];
  const float* bkv        = (const float*)d_in[5];
  const float* Wproj      = (const float*)d_in[6];
  const float* bproj      = (const float*)d_in[7];
  const float* bias_table = (const float*)d_in[8];
  const int*   rel_index  = (const int*)d_in[9];
  float* out = (float*)d_out;

  // workspace: packed f16 weights then f32 gathered bias
  _Float16* WqP    = (_Float16*)d_ws;
  _Float16* WkvP   = WqP  + CH * CH;
  _Float16* WprojP = WkvP + CH * 2 * CH;
  float*    biasG  = (float*)(WprojP + CH * CH);

  swin_prep_kernel<<<128, 256, 0, stream>>>(Wq, Wkv, Wproj, bias_table,
                                            rel_index, WqP, WkvP, WprojP, biasG);

  const size_t smem_bytes =
      (size_t)(4 * NPAD * CH + NPAD * NPAD) * sizeof(_Float16) +
      (size_t)(NPAD * NPAD) * sizeof(float);  // 122880 B
  hipFuncSetAttribute(reinterpret_cast<const void*>(swin_attn_kernel),
                      hipFuncAttributeMaxDynamicSharedMemorySize,
                      (int)smem_bytes);

  const int B_ = in_sizes[0] / (NTOK * CH);  // 8192 windows
  swin_attn_kernel<<<dim3(B_), dim3(256), smem_bytes, stream>>>(
      xq, mask, bq, bkv, bproj, WqP, WkvP, WprojP, biasG, out, B_);
}
